// OAIAttention_5282809774795
// MI455X (gfx1250) — matrix operand order repaired and checked by reading, not compiled, not run on hardware
//
#include <hip/hip_runtime.h>
#include <hip/hip_bf16.h>
#include <math.h>

// ---------------------------------------------------------------------------
// CDNA5 (gfx1250) wave32 WMMA implementation of windowed GQA attention layer.
// LDS tiles live in WMMA fragment-major layout: readers fetch each 32B lane
// fragment with 2x ds_load_b128; writers pack 4-aligned K-quads with
// v_cvt_pk_bf16_f32 pairs and store them with a single ds_store_b64.
// ---------------------------------------------------------------------------

typedef __bf16 bf16_t;
typedef __attribute__((ext_vector_type(16))) __bf16 v16bf;
typedef __attribute__((ext_vector_type(4)))  __bf16 v4bf;
typedef __attribute__((ext_vector_type(8)))  float  v8f;
typedef __attribute__((ext_vector_type(4)))  float  v4f;

#define T_LEN   2048
#define HID     2880
#define NQ      64
#define NKV     8
#define HD      64
#define WIN     128
#define QKV_N   5120      // NQ*HD + 2*NKV*HD
#define Q_SIZE  4096
#define KV_SIZE 512
#define SCALE_F 0.125f    // HD^-0.5

__device__ __forceinline__ v8f wmma_bf16(v16bf a, v16bf b, v8f c) {
  // D = A(16x32 bf16) * B(32x16 bf16) + C(16x16 f32)
  return __builtin_amdgcn_wmma_f32_16x16x32_bf16(
      /*neg_a=*/false, a, /*neg_b=*/false, b,
      /*c_mod=*/(short)0, c, /*reuse_a=*/false, /*reuse_b=*/false);
}

__device__ __forceinline__ v16bf lds_frag(const bf16_t* p) {
  return *(const v16bf*)p;   // 32B aligned contiguous fragment
}
// Pack a 4-aligned K-quad (same lane, consecutive elems) into one b64 store.
__device__ __forceinline__ void store_frag4(bf16_t* dst, v4f v) {
  *(v4bf*)dst = __builtin_convertvector(v, v4bf);
}

// 16-bit A layout: (m,k) -> lane m + (k&16?16:0), elem k&15
__device__ __forceinline__ int a_lane(int m, int k) { return m + ((k & 16) ? 16 : 0); }
__device__ __forceinline__ int a_elem(int k)        { return k & 15; }
// 16-bit B layout used here (contiguous k, not ISA table order): (k,n) -> lane n + (k&16?16:0), elem k&15
__device__ __forceinline__ int b_lane(int n, int k) { return n + ((k & 16) ? 16 : 0); }

// ---------------------------------------------------------------------------
// Generic GEMM: C[M,N] = A[M,K] f32 * B[K,N] f32 + bias, bf16 WMMA compute.
// Block tile 64x64, K-step 32, 128 threads (4 waves).
// ---------------------------------------------------------------------------
#define BM 64
#define BN 64
#define BK 32

__global__ __launch_bounds__(128)
void wmma_gemm_bias(const float* __restrict__ A, const float* __restrict__ B,
                    const float* __restrict__ bias, float* __restrict__ C,
                    int M, int N, int K) {
  // Fragment-major LDS: [tile][lane][16 elems]
  __shared__ alignas(32) bf16_t sAf[4 * 32 * 16];  // 4 row-tiles
  __shared__ alignas(32) bf16_t sBf[4 * 32 * 16];  // 4 col-tiles

  const int tid  = threadIdx.x;
  const int lane = tid & 31;
  const int w    = tid >> 5;
  const int row0 = blockIdx.y * BM;
  const int col0 = blockIdx.x * BN;

  v8f acc[4] = {};

  for (int kk = 0; kk < K; kk += BK) {
    // Stage A (64 rows x 32 k): 16B global load -> packed b64 LDS store.
    #pragma unroll
    for (int seg = 0; seg < 4; ++seg) {
      int idx = tid + seg * 128;             // 512 K-quads
      int r   = idx >> 3;                    // row 0..63
      int c4  = (idx & 7) * 4;               // k quad base
      v4f v = *(const v4f*)&A[(size_t)(row0 + r) * K + kk + c4];
      store_frag4(&sAf[((r >> 4) * 32 + a_lane(r & 15, c4)) * 16 + a_elem(c4)], v);
    }
    // Stage B (32 k x 64 cols): per-thread K-quad down a column (coalesced
    // across lanes), packed b64 LDS store.
    #pragma unroll
    for (int seg = 0; seg < 4; ++seg) {
      int idx = tid + seg * 128;             // 512 K-quads
      int n   = idx & 63;                    // col 0..63
      int k0  = (idx >> 6) * 4;              // k quad base 0..28
      v4f v;
      #pragma unroll
      for (int u = 0; u < 4; ++u)
        v[u] = B[(size_t)(kk + k0 + u) * N + col0 + n];
      store_frag4(&sBf[((n >> 4) * 32 + b_lane(n & 15, k0)) * 16 + (k0 & 15)], v);
    }
    __syncthreads();

    v16bf a = lds_frag(&sAf[(w * 32 + lane) * 16]);
    #pragma unroll
    for (int t = 0; t < 4; ++t) {
      v16bf b = lds_frag(&sBf[(t * 32 + lane) * 16]);
      acc[t] = wmma_bf16(a, b, acc[t]);
    }
    __syncthreads();
  }

  #pragma unroll
  for (int t = 0; t < 4; ++t) {
    #pragma unroll
    for (int r = 0; r < 8; ++r) {
      int row = row0 + w * 16 + (lane >> 4) * 8 + r;  // ISA C/D layout
      int col = col0 + t * 16 + (lane & 15);
      C[(size_t)row * N + col] = acc[t][r] + bias[col];
    }
  }
}

// ---------------------------------------------------------------------------
// YaRN NeoX RoPE applied in place to q and k slices of the qkv buffer.
// ---------------------------------------------------------------------------
__global__ void rope_yarn(float* __restrict__ qkv, const int* __restrict__ positions) {
  const int total = T_LEN * (NQ + NKV) * (HD / 2);
  int idx = blockIdx.x * blockDim.x + threadIdx.x;
  if (idx >= total) return;
  int d    = idx & 31;
  int rest = idx >> 5;
  int head = rest % (NQ + NKV);
  int t    = rest / (NQ + NKV);

  float pf        = __powf(150000.f, (2.f * (float)d) / 64.f);
  float inv_extra = 1.f / pf;
  float inv_inter = 1.f / (32.f * pf);
  const float denom = 2.f * __logf(150000.f);
  float low  = floorf(64.f * __logf(4096.f / (32.f * 6.2831853071f)) / denom);
  float high = ceilf (64.f * __logf(4096.f / ( 1.f * 6.2831853071f)) / denom);
  low  = fmaxf(low, 0.f);
  high = fminf(high, 63.f);
  if (high <= low) high = low + 0.001f;
  float ramp = fminf(fmaxf(((float)d - low) / (high - low), 0.f), 1.f);
  float inv_freq = inv_inter * ramp + inv_extra * (1.f - ramp);
  float mscale   = 0.1f * __logf(32.f) + 1.f;

  float fr = (float)positions[t] * inv_freq;
  float c  = __cosf(fr) * mscale;
  float s  = __sinf(fr) * mscale;

  float* base = (head < NQ)
      ? (qkv + (size_t)t * QKV_N + head * HD)
      : (qkv + (size_t)t * QKV_N + Q_SIZE + (head - NQ) * HD);
  float x1 = base[d];
  float x2 = base[d + 32];
  base[d]      = x1 * c - x2 * s;
  base[d + 32] = x2 * c + x1 * s;
}

// ---------------------------------------------------------------------------
// Sliding-window causal attention with sink logits. One wave per
// (head, 16-query block); window=128 => <=144 live keys, padded to 160.
// Q/K/V/P staged in fragment-major LDS layout; P kept unnormalized and the
// per-row 1/sum applied in the output epilogue.
// ---------------------------------------------------------------------------
#define SMAX 160

__global__ __launch_bounds__(64)
void attn_swin(const float* __restrict__ qkv, const float* __restrict__ sinks,
               float* __restrict__ attn) {
  __shared__ alignas(32) bf16_t qf[2][2 * 32 * 16];  // 2 K-frags of Q
  __shared__ alignas(32) bf16_t kf[2][2 * 32 * 16];  // 2 K-frags of K-tile
  __shared__ alignas(32) bf16_t vf[2][4 * 32 * 16];  // 4 N-frags of V-chunk
  __shared__ alignas(16) float  sbuf[2][16 * SMAX];  // score strip
  __shared__ alignas(32) bf16_t pfb[2][5 * 32 * 16]; // 5 chunk A-frags of P
  __shared__ float              rowinv[2][16];       // per-row 1/softmax-sum

  const int lane = threadIdx.x & 31;
  const int w    = threadIdx.x >> 5;
  const int h    = blockIdx.x;
  const int qb   = blockIdx.y * 2 + w;
  const int kvh  = h >> 3;          // GQA: 8 query heads per kv head
  const int q0   = qb * 16;

  bf16_t* qW = qf[w];
  bf16_t* kW = kf[w];
  bf16_t* vW = vf[w];
  float*  sB = sbuf[w];
  bf16_t* pW = pfb[w];
  float*  rI = rowinv[w];

  // ---- stage Q (16 x 64) into A-frag layout ----
  #pragma unroll
  for (int seg = 0; seg < 8; ++seg) {
    int idx = lane + seg * 32;                  // 256 K-quads
    int r   = idx >> 4;                         // query row 0..15
    int d4  = (idx & 15) * 4;
    v4f v = *(const v4f*)&qkv[(size_t)(q0 + r) * QKV_N + h * HD + d4];
    int f = d4 >> 5, kp = d4 & 31;
    store_frag4(&qW[(f * 32 + a_lane(r, kp)) * 16 + a_elem(kp)], v);
  }
  asm volatile("s_wait_dscnt 0x0" ::: "memory");

  v16bf aq0 = lds_frag(&qW[(0 * 32 + lane) * 16]);
  v16bf aq1 = lds_frag(&qW[(1 * 32 + lane) * 16]);

  const int kt_lo  = (qb > 8) ? (qb - 8) : 0;
  const int ntiles = qb - kt_lo + 1;            // <= 9 key tiles of 16
  const int nk     = ntiles * 16;

  // ---- S = Q * K^T (scaled, window-masked) into LDS strip ----
  for (int kt = kt_lo; kt <= qb; ++kt) {
    #pragma unroll
    for (int seg = 0; seg < 8; ++seg) {
      int idx = lane + seg * 32;                // 256 K-quads (dims)
      int r   = idx >> 4;                       // key row 0..15
      int d4  = (idx & 15) * 4;
      v4f v = *(const v4f*)
          &qkv[(size_t)(kt * 16 + r) * QKV_N + Q_SIZE + kvh * HD + d4];
      int f = d4 >> 5, kp = d4 & 31;
      store_frag4(&kW[(f * 32 + b_lane(r, kp)) * 16 + (kp & 15)], v);
    }
    asm volatile("s_wait_dscnt 0x0" ::: "memory");

    v16bf bk0 = lds_frag(&kW[(0 * 32 + lane) * 16]);
    v16bf bk1 = lds_frag(&kW[(1 * 32 + lane) * 16]);
    v8f sf = {};
    sf = wmma_bf16(aq0, bk0, sf);
    sf = wmma_bf16(aq1, bk1, sf);

    int col0 = (kt - kt_lo) * 16;
    #pragma unroll
    for (int r = 0; r < 8; ++r) {
      int mi = (lane >> 4) * 8 + r;
      int qi = q0 + mi;
      int kj = kt * 16 + (lane & 15);
      float v = ((kj <= qi) && ((qi - kj) < WIN)) ? sf[r] * SCALE_F : -1e30f;
      sB[mi * SMAX + col0 + (lane & 15)] = v;
    }
  }
  asm volatile("s_wait_dscnt 0x0" ::: "memory");

  // ---- softmax (sink in max & denominator); store unnormalized bf16 P ----
  const int nchunks = (ntiles + 1) >> 1;        // 32-key chunks for PV
  const int nkp     = nchunks * 32;
  if (lane < 16) {
    const float* srow = &sB[lane * SMAX];
    float snk = sinks[h];
    float mx  = snk;
    for (int j = 0; j < nk; j += 4) {
      v4f q = *(const v4f*)&srow[j];
      mx = fmaxf(mx, fmaxf(fmaxf(q[0], q[1]), fmaxf(q[2], q[3])));
    }
    float sum = __expf(snk - mx);
    for (int j = 0; j < nkp; j += 4) {
      v4f p = {0.f, 0.f, 0.f, 0.f};
      if (j < nk) {
        v4f q = *(const v4f*)&srow[j];
        p[0] = __expf(q[0] - mx); p[1] = __expf(q[1] - mx);
        p[2] = __expf(q[2] - mx); p[3] = __expf(q[3] - mx);
        sum += (p[0] + p[1]) + (p[2] + p[3]);
      }
      int kp = j & 31;
      store_frag4(&pW[((j >> 5) * 32 + a_lane(lane, kp)) * 16 + a_elem(kp)], p);
    }
    rI[lane] = 1.f / sum;
  }
  asm volatile("s_wait_dscnt 0x0" ::: "memory");

  // ---- O = P * V (unnormalized) ----
  v8f o[4] = {};
  const int last_key = q0 + 15;
  for (int c = 0; c < nchunks; ++c) {
    #pragma unroll
    for (int seg = 0; seg < 16; ++seg) {
      int idx = lane + seg * 32;                // 512 K-quads (keys)
      int d   = idx & 63;                       // out dim
      int r0  = (idx >> 6) * 4;                 // key quad base 0..28
      v4f v;
      #pragma unroll
      for (int u = 0; u < 4; ++u) {
        int key = kt_lo * 16 + c * 32 + r0 + u;
        key = (key > last_key) ? last_key : key;   // pad keys carry p==0
        v[u] = qkv[(size_t)key * QKV_N + Q_SIZE + KV_SIZE + kvh * HD + d];
      }
      store_frag4(&vW[((d >> 4) * 32 + b_lane(d & 15, r0)) * 16 + (r0 & 15)], v);
    }
    asm volatile("s_wait_dscnt 0x0" ::: "memory");

    v16bf ap = lds_frag(&pW[(c * 32 + lane) * 16]);
    #pragma unroll
    for (int t = 0; t < 4; ++t) {
      v16bf bv = lds_frag(&vW[(t * 32 + lane) * 16]);
      o[t] = wmma_bf16(ap, bv, o[t]);
    }
  }

  // ---- epilogue: apply per-row 1/sum, store ----
  float rinv[8];
  #pragma unroll
  for (int r = 0; r < 8; ++r) rinv[r] = rI[(lane >> 4) * 8 + r];
  #pragma unroll
  for (int t = 0; t < 4; ++t) {
    #pragma unroll
    for (int r = 0; r < 8; ++r) {
      int row = q0 + (lane >> 4) * 8 + r;
      int col = h * HD + t * 16 + (lane & 15);
      attn[(size_t)row * Q_SIZE + col] = o[t][r] * rinv[r];
    }
  }
}

// ---------------------------------------------------------------------------
extern "C" void kernel_launch(void* const* d_in, const int* in_sizes, int n_in,
                              void* d_out, int out_size, void* d_ws, size_t ws_size,
                              hipStream_t stream) {
  (void)in_sizes; (void)n_in; (void)out_size; (void)ws_size;

  const float* hidden    = (const float*)d_in[0];
  const int*   positions = (const int*)  d_in[1];
  const float* w_qkv     = (const float*)d_in[2];
  const float* b_qkv     = (const float*)d_in[3];
  const float* w_o       = (const float*)d_in[4];
  const float* b_o       = (const float*)d_in[5];
  const float* sinks     = (const float*)d_in[6];

  float* qkv  = (float*)d_ws;                        // T_LEN * 5120 f32
  float* attn = qkv + (size_t)T_LEN * QKV_N;         // T_LEN * 4096 f32

  // 1) QKV projection: [2048,2880] x [2880,5120] + bias
  wmma_gemm_bias<<<dim3(QKV_N / BN, T_LEN / BM), 128, 0, stream>>>(
      hidden, w_qkv, b_qkv, qkv, T_LEN, QKV_N, HID);

  // 2) YaRN RoPE on q and k in place
  {
    int total = T_LEN * (NQ + NKV) * (HD / 2);
    rope_yarn<<<(total + 255) / 256, 256, 0, stream>>>(qkv, positions);
  }

  // 3) Sliding-window attention with sinks: one wave per (head, 16 queries)
  attn_swin<<<dim3(NQ, T_LEN / 32), 64, 0, stream>>>(qkv, sinks, attn);

  // 4) Output projection: [2048,4096] x [4096,2880] + bias
  wmma_gemm_bias<<<dim3(HID / BN, T_LEN / BM), 128, 0, stream>>>(
      attn, w_o, b_o, (float*)d_out, T_LEN, HID, Q_SIZE);
}
